// ContrastHead_83416854823320
// MI455X (gfx1250) — matrix operand order repaired and checked by reading, not compiled, not run on hardware
//
#include <hip/hip_runtime.h>
#include <hip/hip_bf16.h>
#include <math.h>

typedef __attribute__((ext_vector_type(16))) _Float16 v16h;
typedef __attribute__((ext_vector_type(8)))  float    v8f;

#define KNB 16          // neighbors per point
#define CF  32          // feature channels
#define ROWSTRIDE 40    // padded LDS row stride in halves (breaks bank conflicts)
#define TEMP_INV 10.0f  // 1 / TEMPERATURE
#define EPSF 1e-8f
#define WEIGHTF 0.1f
#define WAVES_PER_BLOCK 8

__global__ void __launch_bounds__(256)
contrast_head_kernel(const float* __restrict__ features,
                     const int*   __restrict__ labels,
                     const long long* __restrict__ nbidx,
                     float* __restrict__ accum,
                     int n)
{
    __shared__ _Float16 s_diff[WAVES_PER_BLOCK][KNB * ROWSTRIDE];
    __shared__ float    s_cf[WAVES_PER_BLOCK][CF];

    const int lane = threadIdx.x & 31;
    const int wave = threadIdx.x >> 5;
    const int gw   = blockIdx.x * WAVES_PER_BLOCK + wave;
    const int nw   = gridDim.x * WAVES_PER_BLOCK;

    // half-row gather mapping: lane L covers neighbor row L>>1, cols (L&1)*16 .. +15
    const int r  = lane >> 1;
    const int cb = (lane & 1) * 16;

    float loss_acc = 0.0f;
    float cnt_acc  = 0.0f;

    for (int p = gw; p < n; p += nw) {
        // ---- stage center feature row (coalesced 128B) ----
        s_cf[wave][lane] = features[(size_t)p * CF + lane];
        __builtin_amdgcn_wave_barrier();

        // ---- load the 16 neighbor indices once (lanes 0..15) ----
        long long nkl = 0;
        bool pm = false;
        const int lc = labels[p];
        if (lane < KNB) {
            nkl = nbidx[(size_t)p * KNB + lane];
            pm  = (labels[nkl] == lc);               // posmask
        }

        // broadcast row index to the two gather lanes of each row
        const long long nk = __shfl(nkl, r, 32);

        // ---- gather neighbor half-row, write f16 diff tile to LDS ----
        const float4* src = (const float4*)(features + (size_t)nk * CF + cb);
        float4 f0 = src[0], f1 = src[1], f2 = src[2], f3 = src[3];
        float fv[16] = { f0.x, f0.y, f0.z, f0.w,
                         f1.x, f1.y, f1.z, f1.w,
                         f2.x, f2.y, f2.z, f2.w,
                         f3.x, f3.y, f3.z, f3.w };
        const float* cf = &s_cf[wave][cb];
        _Float16* drow = &s_diff[wave][r * ROWSTRIDE + cb];
        #pragma unroll
        for (int t = 0; t < 16; ++t)
            drow[t] = (_Float16)(cf[t] - fv[t]);
        __builtin_amdgcn_wave_barrier();

        // ---- load WMMA fragments per 16-bit A / B layouts (ISA 7.12.2) ----
        const int mrow  = (lane < 16) ? lane : (lane - 16);
        const int abase = (lane < 16) ? 0 : 8;    // A: K {0-7,16-23} vs {8-15,24-31}
    const int bbase = (lane < 16) ? 0 : 8;    // B: diff[n] K {0-7,16-23} vs {8-15,24-31}
        const _Float16* rowp = &s_diff[wave][mrow * ROWSTRIDE];
        v16h a, b;
        #pragma unroll
        for (int t = 0; t < 8; ++t) {
            a[t]     = rowp[abase + t];
            a[t + 8] = rowp[abase + 16 + t];
        }
        #pragma unroll
        for (int t = 0; t < 8; ++t) {
            b[t]     = rowp[bbase + t];
            b[t + 8] = rowp[bbase + 16 + t];
        }

        v8f c = {0.f, 0.f, 0.f, 0.f, 0.f, 0.f, 0.f, 0.f};
        // Gram = diff * diff^T ; diagonal = squared L2 distances
        v8f d = __builtin_amdgcn_wmma_f32_16x16x32_f16(
                    false, a, false, b, (short)0, c, false, false);

        // ---- branch-free diagonal extraction ----
        // diag[m] sits in VGPR (m&7) of lane m (m<8) or lane m+16 (m>=8).
        // Every lane selects d[lane&7] (cndmask tree), then lane k<16 pulls it
        // from source lane (k<8 ? k : k+16) with one ds_bpermute.
        const int s = lane & 7;
        float a0 = (s & 1) ? d[1] : d[0];
        float a1 = (s & 1) ? d[3] : d[2];
        float a2 = (s & 1) ? d[5] : d[4];
        float a3 = (s & 1) ? d[7] : d[6];
        float b0 = (s & 2) ? a1 : a0;
        float b1 = (s & 2) ? a3 : a2;
        float dsel = (s & 4) ? b1 : b0;
        const int srcLane = (lane < 8) ? lane : (lane + 16);
        float dist2 = __shfl(dsel, srcLane, 32);

        // ---- softmax-contrast loss across the 16 neighbors, branch-free ----
        // Upper 16 lanes get a huge distance; the xor butterflies stay within each
        // 16-lane half and never mix with lanes 0..15, so no predication is needed anywhere.
        float d2 = (lane < KNB) ? dist2 : 1.0e30f;
        float nd = -__builtin_amdgcn_sqrtf(d2 + EPSF);   // raw v_sqrt_f32

        float m = nd;                       // max over the 16-lane group
        #pragma unroll
        for (int sh = 8; sh >= 1; sh >>= 1)
            m = fmaxf(m, __shfl_xor(m, sh, 32));

        float e  = __expf((nd - m) * TEMP_INV);          // 1.0 on lanes >= 16
        float ep = pm ? e : 0.f;

        float neg = e, pos = ep;            // sums over the 16-lane group
        #pragma unroll
        for (int sh = 8; sh >= 1; sh >>= 1) {
            neg += __shfl_xor(neg, sh, 32);
            pos += __shfl_xor(pos, sh, 32);
        }

        // point_mask: 0 < posmask_cnt < K  (ballot + mask compare, no butterfly)
        const unsigned pmask = (unsigned)(__ballot(pm) & 0xFFFFull);
        const bool valid = (pmask != 0u) && (pmask != 0xFFFFu);
        const float w = (lane == 0 && valid) ? 1.0f : 0.0f;
        loss_acc += w * (-__logf(pos / neg + EPSF));
        cnt_acc  += w;
    }

    if (lane == 0) {
        atomicAdd(&accum[0], loss_acc);
        atomicAdd(&accum[1], cnt_acc);
    }
}

__global__ void contrast_init_kernel(float* ws) {
    if (threadIdx.x < 2) ws[threadIdx.x] = 0.0f;
}

__global__ void contrast_final_kernel(const float* __restrict__ ws,
                                      float* __restrict__ out) {
    if (threadIdx.x == 0)
        out[0] = WEIGHTF * ws[0] / fmaxf(ws[1], 1.0f);
}

extern "C" void kernel_launch(void* const* d_in, const int* in_sizes, int n_in,
                              void* d_out, int out_size, void* d_ws, size_t ws_size,
                              hipStream_t stream) {
    const float*     features = (const float*)d_in[0];
    const int*       labels   = (const int*)d_in[1];
    const long long* nbidx    = (const long long*)d_in[2];  // int64 in reference
    float* out = (float*)d_out;
    float* ws  = (float*)d_ws;

    const int n = in_sizes[1];   // labels: [N]

    contrast_init_kernel<<<1, 32, 0, stream>>>(ws);

    // one wave per point, grid-stride; 8192 waves -> ~25 points/wave
    const int blocks = 1024;
    contrast_head_kernel<<<blocks, 256, 0, stream>>>(features, labels, nbidx, ws, n);

    contrast_final_kernel<<<1, 32, 0, stream>>>(ws, out);
}
